// MultiHeadSelfAttention_53575422051008
// MI455X (gfx1250) — matrix operand order repaired and checked by reading, not compiled, not run on hardware
//
#include <hip/hip_runtime.h>
#include <hip/hip_bf16.h>

// ---------------------------------------------------------------------------
// MHA forward for MI455X (gfx1250): bf16 WMMA, f32 accumulate, async-to-LDS
// double-buffered weight staging, flash attention with causal mask + RoPE.
// B=2, S=2048, D=1024, H=16, hd=64.
// ---------------------------------------------------------------------------

#define S_LEN   2048
#define DMODEL  1024
#define NHEADS  16
#define HDIM    64
#define BATCH   2
#define MROWS   (BATCH * S_LEN)   // 4096
#define LDB     40                // LDS B-tile row stride (ushorts): 32 + 8 pad

typedef __attribute__((ext_vector_type(16))) __bf16 v16bf;
typedef __attribute__((ext_vector_type(8)))  float  v8f;

union Frag16 { v16bf v; unsigned int u[8]; };

static __device__ __forceinline__ unsigned short f2bf(float f) {
  unsigned int u = __float_as_uint(f);
  u += 0x7fffu + ((u >> 16) & 1u);          // round-to-nearest-even
  return (unsigned short)(u >> 16);
}

static __device__ __forceinline__ unsigned int ld_u32(const unsigned short* p) {
  return *(const unsigned int*)p;
}

// A-fragment (16x32 bf16, M x K), row-major source, ld in elements.
// ISA: lane -> row M = lane&15; lanes 0-15: VGPR0-3=K0..7, VGPR4-7=K16..23;
//      lanes 16-31: VGPR0-3=K8..15, VGPR4-7=K24..31. Two bf16 per VGPR.
static __device__ __forceinline__ Frag16 load_a(const unsigned short* base, int ld) {
  const int lane = threadIdx.x & 31;
  const int row  = lane & 15;
  const int half = lane >> 4;
  Frag16 F;
#pragma unroll
  for (int v = 0; v < 4; ++v)
    F.u[v] = ld_u32(base + row * ld + half * 8 + 2 * v);
#pragma unroll
  for (int v = 0; v < 4; ++v)
    F.u[4 + v] = ld_u32(base + row * ld + 16 + half * 8 + 2 * v);
  return F;
}

// B-fragment (32x16 bf16, K x N) from a transposed (N x K, K-contiguous)
// source. ISA: lane -> column N = lane&15; lanes 0-15: VGPR0-3=K0..7,
// VGPR4-7=K16..23; lanes 16-31: VGPR0-3=K8..15, VGPR4-7=K24..31.
static __device__ __forceinline__ Frag16 load_bt(const unsigned short* base, int ld) {
  const int lane = threadIdx.x & 31;
  const int col  = lane & 15;
  const int kb   = (lane >> 4) * 8;
  Frag16 F;
#pragma unroll
  for (int v = 0; v < 4; ++v)
    F.u[v] = ld_u32(base + col * ld + kb + 2 * v);
#pragma unroll
  for (int v = 0; v < 4; ++v)
    F.u[4 + v] = ld_u32(base + col * ld + 16 + kb + 2 * v);
  return F;
}

static __device__ __forceinline__ v8f wmma_bf16(const Frag16& a, const Frag16& b, v8f c) {
  return __builtin_amdgcn_wmma_f32_16x16x32_bf16(false, a.v, false, b.v,
                                                 (short)0, c, false, false);
}

// Async-copy a 64(N) x 32(K) bf16 tile (N-major, K-contiguous source rows of
// 64 B) into LDS with padded rows. Each wave copies 16 rows: 2 instructions
// x 32 lanes x 16 B. Tracked by ASYNCcnt.
static __device__ __forceinline__ void async_copy_b_tile(
    const unsigned short* __restrict__ src,  // = W + n0*DMODEL + k0
    unsigned short* lds_base,                // tile buffer (alignas(16))
    int wave, int lane) {
  const int seg = lane & 3;        // 16-byte segment within a 64-byte row
  const int r0  = lane >> 2;       // 0..7
#pragma unroll
  for (int i = 0; i < 2; ++i) {
    const int row = 16 * wave + 8 * i + r0;
    const unsigned short* g = src + (size_t)row * DMODEL + seg * 8;
    unsigned int l = (unsigned int)(unsigned long long)(lds_base + row * LDB + seg * 8);
    asm volatile("global_load_async_to_lds_b128 %0, %1, off"
                 :: "v"(l), "v"(g) : "memory");
  }
}

// ---------------------------------------------------------------------------
// f32 -> bf16 conversion (grid-stride)
// ---------------------------------------------------------------------------
__global__ void cvt_kernel(const float* __restrict__ src,
                           unsigned short* __restrict__ dst, int n) {
  for (int i = blockIdx.x * blockDim.x + threadIdx.x; i < n;
       i += gridDim.x * blockDim.x)
    dst[i] = f2bf(src[i]);
}

// ---------------------------------------------------------------------------
// QKV projection + RoPE + head-major relayout.
// blockIdx.z: 0 -> Q (rope), 1 -> K (rope), 2 -> V (transposed store).
// Block = 4 waves, tile 128(M) x 64(N); B-tile double-buffered in LDS via
// global_load_async_to_lds_b128; 8 WMMA per K-step.
// ---------------------------------------------------------------------------
__global__ __launch_bounds__(128) void qkv_rope_kernel(
    const unsigned short* __restrict__ xb,
    const unsigned short* __restrict__ wqb,
    const unsigned short* __restrict__ wkb,
    const unsigned short* __restrict__ wvb,
    unsigned short* __restrict__ Qb,     // [b,h,s,d]
    unsigned short* __restrict__ Kb,     // [b,h,s,d]
    unsigned short* __restrict__ Vtb) {  // [b,h,d,s]
  __shared__ __align__(16) unsigned short tb[2][64 * LDB];
  const int wave = threadIdx.x >> 5;
  const int lane = threadIdx.x & 31;
  const int half = lane >> 4, ln = lane & 15;
  const int m0 = blockIdx.x * 128 + wave * 32;
  const int n0 = blockIdx.y * 64;
  const int which = blockIdx.z;
  const unsigned short* W = (which == 0) ? wqb : (which == 1) ? wkb : wvb;

  v8f acc[2][4];
#pragma unroll
  for (int t = 0; t < 2; ++t)
#pragma unroll
    for (int f = 0; f < 4; ++f)
      acc[t][f] = (v8f){0.f,0.f,0.f,0.f,0.f,0.f,0.f,0.f};

  async_copy_b_tile(W + (size_t)n0 * DMODEL, tb[0], wave, lane);

  for (int k0 = 0; k0 < DMODEL; k0 += 32) {
    const int buf = (k0 >> 5) & 1;
    __syncthreads();                      // WAR: tb[buf^1] readers finished
    if (k0 + 32 < DMODEL) {
      async_copy_b_tile(W + (size_t)n0 * DMODEL + k0 + 32, tb[buf ^ 1], wave, lane);
      asm volatile("s_wait_asynccnt 0x2" ::: "memory");  // tile 'buf' landed
    } else {
      asm volatile("s_wait_asynccnt 0x0" ::: "memory");
    }
    __syncthreads();                      // all waves' quarters visible

    __builtin_prefetch(xb + (size_t)m0 * DMODEL + k0 + 64, 0, 1);
    Frag16 a0 = load_a(xb + (size_t)m0 * DMODEL + k0, DMODEL);
    Frag16 a1 = load_a(xb + (size_t)(m0 + 16) * DMODEL + k0, DMODEL);
#pragma unroll
    for (int f = 0; f < 4; ++f) {
      Frag16 b = load_bt(tb[buf] + 16 * f * LDB, LDB);
      acc[0][f] = wmma_bf16(a0, b, acc[0][f]);
      acc[1][f] = wmma_bf16(a1, b, acc[1][f]);
    }
  }

  // Epilogue: RoPE (for Q/K) + store in attention-friendly layouts.
#pragma unroll
  for (int t = 0; t < 2; ++t)
#pragma unroll
    for (int f = 0; f < 4; ++f)
#pragma unroll
      for (int r = 0; r < 8; ++r) {
        const int m = m0 + 16 * t + 8 * half + r;   // global row (b*S + s)
        const int n = n0 + 16 * f + ln;             // global col (h*64 + d)
        const int b = m >> 11, s = m & (S_LEN - 1);
        const int h = n >> 6,  d = n & 63;
        const int bh = b * NHEADS + h;
        const float val = acc[t][f][r];
        // rotate_half partner lives in the adjacent lane (col n^1)
        const float partner = __shfl_xor(val, 1, 32);
        if (which == 2) {
          Vtb[(size_t)bh * HDIM * S_LEN + (size_t)d * S_LEN + s] = f2bf(val);
        } else {
          // freq = 10000^{-(d&~1)/64}; ln(10000)/64 = 0.14391156...
          const float freq = __expf(-0.14391156f * (float)(d & 62));
          const float ang = (float)s * freq;
          float sn, cs;
          __sincosf(ang, &sn, &cs);
          const float res = (d & 1) ? (val * cs + partner * sn)
                                    : (val * cs - partner * sn);
          unsigned short* dst = (which == 0) ? Qb : Kb;
          dst[(size_t)bh * S_LEN * HDIM + (size_t)s * HDIM + d] = f2bf(res);
        }
      }
}

// ---------------------------------------------------------------------------
// Causal flash attention. Block = 4 waves; wave owns 16 queries of a 64-query
// tile. Keys processed in tiles of 32. Per tile: 4 score WMMA + 4 PV WMMA.
// P is re-laid-out C->A through per-wave LDS.
// ---------------------------------------------------------------------------
__global__ __launch_bounds__(128) void attn_kernel(
    const unsigned short* __restrict__ Qb,
    const unsigned short* __restrict__ Kb,
    const unsigned short* __restrict__ Vtb,
    unsigned short* __restrict__ Mb) {   // merged [b,s,h*64+d] bf16
  __shared__ __align__(16) unsigned short pbuf[4][16 * 32];
  const int wave = threadIdx.x >> 5;
  const int lane = threadIdx.x & 31;
  const int half = lane >> 4, ln = lane & 15;
  const int qt = blockIdx.x;                 // query tile of 64
  const int h = blockIdx.y, b = blockIdx.z;
  const int bh = b * NHEADS + h;
  const unsigned short* Qh = Qb  + (size_t)bh * S_LEN * HDIM;
  const unsigned short* Kh = Kb  + (size_t)bh * S_LEN * HDIM;
  const unsigned short* Vh = Vtb + (size_t)bh * HDIM * S_LEN;
  const int q0 = qt * 64 + wave * 16;

  const Frag16 aq0 = load_a(Qh + (size_t)q0 * HDIM + 0,  HDIM);
  const Frag16 aq1 = load_a(Qh + (size_t)q0 * HDIM + 32, HDIM);

  v8f o[4];
#pragma unroll
  for (int f = 0; f < 4; ++f) o[f] = (v8f){0.f,0.f,0.f,0.f,0.f,0.f,0.f,0.f};
  float mrow[8], lrow[8];
#pragma unroll
  for (int r = 0; r < 8; ++r) { mrow[r] = -1e30f; lrow[r] = 0.f; }

  const int nk = qt * 64 + 64;               // uniform causal bound per block
  for (int kt = 0; kt < nk; kt += 32) {
    v8f s0 = (v8f){0.f,0.f,0.f,0.f,0.f,0.f,0.f,0.f};
    v8f s1 = s0;
    {
      Frag16 bk = load_bt(Kh + (size_t)kt * HDIM + 0, HDIM);
      s0 = wmma_bf16(aq0, bk, s0);
      bk = load_bt(Kh + (size_t)kt * HDIM + 32, HDIM);
      s0 = wmma_bf16(aq1, bk, s0);
      bk = load_bt(Kh + (size_t)(kt + 16) * HDIM + 0, HDIM);
      s1 = wmma_bf16(aq0, bk, s1);
      bk = load_bt(Kh + (size_t)(kt + 16) * HDIM + 32, HDIM);
      s1 = wmma_bf16(aq1, bk, s1);
    }
    float p0[8], p1[8], scl[8];
#pragma unroll
    for (int r = 0; r < 8; ++r) {
      const int q = q0 + 8 * half + r;
      float v0 = s0[r] * 0.125f;             // 1/sqrt(64)
      float v1 = s1[r] * 0.125f;
      if (kt + ln > q)      v0 = -1e30f;     // causal mask
      if (kt + 16 + ln > q) v1 = -1e30f;
      float mx = fmaxf(v0, v1);
#pragma unroll
      for (int off = 1; off < 16; off <<= 1)
        mx = fmaxf(mx, __shfl_xor(mx, off, 32));
      const float nm = fmaxf(mrow[r], mx);
      const float sc = __expf(mrow[r] - nm);
      const float e0 = __expf(v0 - nm);
      const float e1 = __expf(v1 - nm);
      float sum = e0 + e1;
#pragma unroll
      for (int off = 1; off < 16; off <<= 1)
        sum += __shfl_xor(sum, off, 32);
      lrow[r] = lrow[r] * sc + sum;
      mrow[r] = nm;
      scl[r] = sc; p0[r] = e0; p1[r] = e1;
    }
#pragma unroll
    for (int f = 0; f < 4; ++f)
#pragma unroll
      for (int r = 0; r < 8; ++r) o[f][r] *= scl[r];

    // C-layout P -> LDS (row-major 16x32 bf16) -> A-fragment.
    unsigned short* pw = pbuf[wave];
#pragma unroll
    for (int r = 0; r < 8; ++r) {
      const int row = 8 * half + r;
      pw[row * 32 + ln]      = f2bf(p0[r]);
      pw[row * 32 + 16 + ln] = f2bf(p1[r]);
    }
    __syncthreads();
    const Frag16 aP = load_a(pw, 32);
    __syncthreads();

#pragma unroll
    for (int f = 0; f < 4; ++f) {
      Frag16 bv = load_bt(Vh + (size_t)(16 * f) * S_LEN + kt, S_LEN);
      o[f] = wmma_bf16(aP, bv, o[f]);
    }
  }

#pragma unroll
  for (int r = 0; r < 8; ++r) {
    const float inv = 1.0f / lrow[r];
    const int q = q0 + 8 * half + r;
#pragma unroll
    for (int f = 0; f < 4; ++f) {
      const int d = 16 * f + ln;
      Mb[((size_t)b * S_LEN + q) * DMODEL + h * HDIM + d] = f2bf(o[f][r] * inv);
    }
  }
}

// ---------------------------------------------------------------------------
// Output projection: out = merged @ wo^T (f32 result), async-staged B tiles.
// ---------------------------------------------------------------------------
__global__ __launch_bounds__(128) void out_proj_kernel(
    const unsigned short* __restrict__ Mb,
    const unsigned short* __restrict__ wob,
    float* __restrict__ out) {
  __shared__ __align__(16) unsigned short tb[2][64 * LDB];
  const int wave = threadIdx.x >> 5;
  const int lane = threadIdx.x & 31;
  const int half = lane >> 4, ln = lane & 15;
  const int m0 = blockIdx.x * 128 + wave * 32;
  const int n0 = blockIdx.y * 64;

  v8f acc[2][4];
#pragma unroll
  for (int t = 0; t < 2; ++t)
#pragma unroll
    for (int f = 0; f < 4; ++f)
      acc[t][f] = (v8f){0.f,0.f,0.f,0.f,0.f,0.f,0.f,0.f};

  async_copy_b_tile(wob + (size_t)n0 * DMODEL, tb[0], wave, lane);

  for (int k0 = 0; k0 < DMODEL; k0 += 32) {
    const int buf = (k0 >> 5) & 1;
    __syncthreads();
    if (k0 + 32 < DMODEL) {
      async_copy_b_tile(wob + (size_t)n0 * DMODEL + k0 + 32, tb[buf ^ 1], wave, lane);
      asm volatile("s_wait_asynccnt 0x2" ::: "memory");
    } else {
      asm volatile("s_wait_asynccnt 0x0" ::: "memory");
    }
    __syncthreads();

    __builtin_prefetch(Mb + (size_t)m0 * DMODEL + k0 + 64, 0, 1);
    Frag16 a0 = load_a(Mb + (size_t)m0 * DMODEL + k0, DMODEL);
    Frag16 a1 = load_a(Mb + (size_t)(m0 + 16) * DMODEL + k0, DMODEL);
#pragma unroll
    for (int f = 0; f < 4; ++f) {
      Frag16 b = load_bt(tb[buf] + 16 * f * LDB, LDB);
      acc[0][f] = wmma_bf16(a0, b, acc[0][f]);
      acc[1][f] = wmma_bf16(a1, b, acc[1][f]);
    }
  }

#pragma unroll
  for (int t = 0; t < 2; ++t)
#pragma unroll
    for (int f = 0; f < 4; ++f)
#pragma unroll
      for (int r = 0; r < 8; ++r) {
        const int m = m0 + 16 * t + 8 * half + r;
        const int n = n0 + 16 * f + ln;
        out[(size_t)m * DMODEL + n] = acc[t][f][r];
      }
}

// ---------------------------------------------------------------------------
extern "C" void kernel_launch(void* const* d_in, const int* in_sizes, int n_in,
                              void* d_out, int out_size, void* d_ws, size_t ws_size,
                              hipStream_t stream) {
  const float* x  = (const float*)d_in[0];
  const float* wq = (const float*)d_in[1];
  const float* wk = (const float*)d_in[2];
  const float* wv = (const float*)d_in[3];
  const float* wo = (const float*)d_in[4];
  float* out = (float*)d_out;

  // Workspace layout (bf16 elements). Total = 5*4M + 4*1M = 24M elems = 48 MB.
  unsigned short* ws = (unsigned short*)d_ws;
  const size_t XB = (size_t)MROWS * DMODEL;   // 4194304
  const size_t WN = (size_t)DMODEL * DMODEL;  // 1048576
  unsigned short* xb  = ws;
  unsigned short* wqb = xb  + XB;
  unsigned short* wkb = wqb + WN;
  unsigned short* wvb = wkb + WN;
  unsigned short* wob = wvb + WN;
  unsigned short* Qb  = wob + WN;
  unsigned short* Kb  = Qb  + XB;
  unsigned short* Vtb = Kb  + XB;
  unsigned short* Mb  = Vtb + XB;

  cvt_kernel<<<1024, 256, 0, stream>>>(x,  xb,  (int)XB);
  cvt_kernel<<<512,  256, 0, stream>>>(wq, wqb, (int)WN);
  cvt_kernel<<<512,  256, 0, stream>>>(wk, wkb, (int)WN);
  cvt_kernel<<<512,  256, 0, stream>>>(wv, wvb, (int)WN);
  cvt_kernel<<<512,  256, 0, stream>>>(wo, wob, (int)WN);

  qkv_rope_kernel<<<dim3(MROWS / 128, DMODEL / 64, 3), 128, 0, stream>>>(
      xb, wqb, wkb, wvb, Qb, Kb, Vtb);

  attn_kernel<<<dim3(S_LEN / 64, NHEADS, BATCH), 128, 0, stream>>>(
      Qb, Kb, Vtb, Mb);

  out_proj_kernel<<<dim3(MROWS / 128, DMODEL / 64), 128, 0, stream>>>(
      Mb, wob, out);
}
